// CRF_67963562492335
// MI455X (gfx1250) — matrix operand order repaired and checked by reading, not compiled, not run on hardware
//
#include <hip/hip_runtime.h>

// CRF loss for MI455X (gfx1250, wave32).
// Forward algorithm rewritten as scaled matrix recurrence:
//   q' = (q * exp(trans)) .* exp(em);  m = rowmax(q');  q = q'/m;  c += log(m)
// computed transposed (D = E^T x q^T) with v_wmma_f32_16x16x32_f16 so the
// f32 accumulator layout feeds the next step's f16 B operand with only
// in-lane cvt_pkrtz packs + ds_swizzle SWAPX16 half-wave swaps.

typedef _Float16 v16h __attribute__((ext_vector_type(16)));
typedef float    v8f  __attribute__((ext_vector_type(8)));
typedef float    f4   __attribute__((ext_vector_type(4)));

constexpr int Bc = 1024;
constexpr int Sc = 512;
constexpr int Tc = 48;

__device__ __forceinline__ unsigned swap16_u(unsigned x) {
  return (unsigned)__builtin_amdgcn_ds_swizzle((int)x, 0x401F); // xor 0x10
}
__device__ __forceinline__ float swap16_f(float x) {
  int i = __builtin_bit_cast(int, x);
  i = __builtin_amdgcn_ds_swizzle(i, 0x401F);
  return __builtin_bit_cast(float, i);
}

union BOp { unsigned u[8]; v16h h; };

__global__ __launch_bounds__(32)
void crf_forward_kernel(const float* __restrict__ em,
                        const float* __restrict__ mask,
                        const float* __restrict__ trans,
                        const float* __restrict__ startt,
                        const float* __restrict__ endt,
                        float* __restrict__ fwd) {
  const int lane = threadIdx.x & 31;
  const int h = (lane >> 4) & 1;     // half-wave: selects j-offset +8
  const int n = lane & 15;           // batch column within group
  const int b = blockIdx.x * 16 + n; // this lane's batch

  // ---- Build resident A operands: A[r][c] = E^T tile, rows j=16r+n, f16 16x32
  // A layout (this code's contiguous k order, matched by B): VGPR v holds K = 32c + 16h + 2v + e
  v16h A[3][2];
#pragma unroll
  for (int r = 0; r < 3; ++r)
#pragma unroll
    for (int c = 0; c < 2; ++c)
#pragma unroll
      for (int v = 0; v < 8; ++v)
#pragma unroll
        for (int e = 0; e < 2; ++e) {
          int K = 32 * c + 16 * h + 2 * v + e;
          float val = (K < Tc) ? __expf(trans[K * Tc + 16 * r + n]) : 0.0f;
          A[r][c][2 * v + e] = (_Float16)val;
        }

  // ---- Init: score0[j] = start[j] + em[b,0,j]; q = exp(score0 - m0); c = m0
  float q[3][8];
  float carry;
  {
    float sc[3][8];
    float m = -3.0e38f;
#pragma unroll
    for (int r = 0; r < 3; ++r) {
      const int j0 = 16 * r + 8 * h;
      const float* p = em + ((size_t)b * Sc) * Tc + j0;
      f4 e0 = *(const f4*)(p);
      f4 e1 = *(const f4*)(p + 4);
#pragma unroll
      for (int k = 0; k < 4; ++k) {
        sc[r][k]     = startt[j0 + k]     + e0[k];
        sc[r][4 + k] = startt[j0 + 4 + k] + e1[k];
      }
#pragma unroll
      for (int k = 0; k < 8; ++k) m = fmaxf(m, sc[r][k]);
    }
    m = fmaxf(m, swap16_f(m)); // row max across both j-halves
#pragma unroll
    for (int r = 0; r < 3; ++r)
#pragma unroll
      for (int k = 0; k < 8; ++k) q[r][k] = __expf(sc[r][k] - m);
    carry = m;
  }

  // ---- Sequential scan over time
  for (int t = 1; t < Sc; ++t) {
    const float mk = mask[(size_t)b * Sc + t];

    // emission factors g = exp(em[b,t,j])
    float gm[3][8];
#pragma unroll
    for (int r = 0; r < 3; ++r) {
      const float* p = em + ((size_t)b * Sc + t) * Tc + 16 * r + 8 * h;
      f4 e0 = *(const f4*)(p);
      f4 e1 = *(const f4*)(p + 4);
#pragma unroll
      for (int k = 0; k < 4; ++k) {
        gm[r][k]     = __expf(e0[k]);
        gm[r][4 + k] = __expf(e1[k]);
      }
    }

    // pack state q -> f16 pairs (j = 16r + 8h + 2p, 2p+1)
    unsigned pk[3][4];
#pragma unroll
    for (int r = 0; r < 3; ++r)
#pragma unroll
      for (int p = 0; p < 4; ++p)
        pk[r][p] = __builtin_bit_cast(unsigned,
                     __builtin_amdgcn_cvt_pkrtz(q[r][2 * p], q[r][2 * p + 1]));

    // assemble B operands (contiguous k order matching A: lanes0-15 K=0..15, lanes16-31 K=16..31)
    BOp B0, B1;
#pragma unroll
    for (int v = 0; v < 4; ++v) {
      unsigned tmix = (h == 0) ? pk[1][v] : pk[0][v];
      unsigned sw   = swap16_u(tmix);          // cross-half exchange
      B0.u[v]     = (h == 0) ? pk[0][v] : sw;  // K=2v,2v+1   | K=16+2v
      B0.u[v + 4] = (h == 0) ? sw : pk[1][v];  // K=8+2v      | K=24+2v
      unsigned sw2 = swap16_u(pk[2][v]);
      B1.u[v]     = (h == 0) ? pk[2][v] : 0u;  // K=32+2v     | pad
      B1.u[v + 4] = (h == 0) ? sw2 : 0u;       // K=40+2v     | pad
    }

    // D[r] = E^T(tile r) x q^T   (two K-chunks accumulated)
    v8f D[3];
#pragma unroll
    for (int r = 0; r < 3; ++r) {
      v8f z = {};
      v8f acc = __builtin_amdgcn_wmma_f32_16x16x32_f16(
          false, A[r][0], false, B0.h, (short)0, z, false, false);
      acc = __builtin_amdgcn_wmma_f32_16x16x32_f16(
          false, A[r][1], false, B1.h, (short)0, acc, false, false);
      D[r] = acc;
    }

    // apply emissions, rescale, masked update
    float qn[3][8];
    float m = -3.0e38f;
#pragma unroll
    for (int r = 0; r < 3; ++r)
#pragma unroll
      for (int k = 0; k < 8; ++k) {
        float val = D[r][k] * gm[r][k];
        qn[r][k] = val;
        m = fmaxf(m, val);
      }
    m = fmaxf(m, swap16_f(m));
    const float inv = 1.0f / m;
    const float lg = __logf(m);
    const bool upd = mk > 0.0f;
#pragma unroll
    for (int r = 0; r < 3; ++r)
#pragma unroll
      for (int k = 0; k < 8; ++k)
        q[r][k] = upd ? qn[r][k] * inv : q[r][k];
    carry = upd ? carry + lg : carry;
  }

  // ---- Finish: fwd[b] = carry + log(sum_j q[j] * exp(end[j]))
  float s = 0.0f;
#pragma unroll
  for (int r = 0; r < 3; ++r) {
    const int j0 = 16 * r + 8 * h;
#pragma unroll
    for (int k = 0; k < 8; ++k) s += q[r][k] * __expf(endt[j0 + k]);
  }
  s += swap16_f(s);
  if (h == 0) fwd[b] = carry + __logf(s);
}

__global__ __launch_bounds__(256)
void crf_gold_kernel(const float* __restrict__ em,
                     const int* __restrict__ tags,
                     const float* __restrict__ mask,
                     const float* __restrict__ trans,
                     const float* __restrict__ startt,
                     const float* __restrict__ endt,
                     float* __restrict__ gold) {
  const int b = blockIdx.x * blockDim.x + threadIdx.x;
  if (b >= Bc) return;
  int prev = tags[(size_t)b * Sc];
  int last = prev;
  float sc = startt[prev] + em[((size_t)b * Sc) * Tc + prev];
  for (int s = 1; s < Sc; ++s) {
    const int tg = tags[(size_t)b * Sc + s];
    const float mk = mask[(size_t)b * Sc + s];
    if (mk > 0.0f) {
      sc += trans[tg * Tc + prev] + em[((size_t)b * Sc + s) * Tc + tg];
      last = tg;
    }
    prev = tg;
  }
  gold[b] = sc + endt[last];
}

__global__ __launch_bounds__(256)
void crf_reduce_kernel(const float* __restrict__ fwd,
                       const float* __restrict__ gold,
                       float* __restrict__ out) {
  __shared__ float sm[256];
  const int tid = threadIdx.x;
  float s = 0.0f;
  for (int i = tid; i < Bc; i += 256) s += fwd[i] - gold[i];
  sm[tid] = s;
  __syncthreads();
  for (int k = 128; k > 0; k >>= 1) {
    if (tid < k) sm[tid] += sm[tid + k];
    __syncthreads();
  }
  if (tid == 0) out[0] = sm[0] / (float)Bc;
}

extern "C" void kernel_launch(void* const* d_in, const int* in_sizes, int n_in,
                              void* d_out, int out_size, void* d_ws, size_t ws_size,
                              hipStream_t stream) {
  const float* em     = (const float*)d_in[0]; // (B,S,T)
  const int*   tags   = (const int*)d_in[1];   // (B,S)
  const float* mask   = (const float*)d_in[2]; // (B,S)
  const float* trans  = (const float*)d_in[3]; // (T,T)
  const float* startt = (const float*)d_in[4]; // (T,)
  const float* endt   = (const float*)d_in[5]; // (T,)
  float* out = (float*)d_out;

  float* fwd  = (float*)d_ws;
  float* gold = fwd + Bc;

  crf_forward_kernel<<<Bc / 16, 32, 0, stream>>>(em, mask, trans, startt, endt, fwd);
  crf_gold_kernel<<<Bc / 256, 256, 0, stream>>>(em, tags, mask, trans, startt, endt, gold);
  crf_reduce_kernel<<<1, 256, 0, stream>>>(fwd, gold, out);
}
